// CNN_binary_relu_55001351193059
// MI455X (gfx1250) — matrix operand order repaired and checked by reading, not compiled, not run on hardware
//
#include <hip/hip_runtime.h>

typedef __attribute__((ext_vector_type(16))) _Float16 v16h;
typedef __attribute__((ext_vector_type(8)))  _Float16 v8h;
typedef __attribute__((ext_vector_type(8)))  float    v8f;
typedef __attribute__((ext_vector_type(4)))  unsigned int u32x4;
typedef __attribute__((ext_vector_type(8)))  int      i32x8;
typedef __attribute__((ext_vector_type(4)))  int      i32x4;

#define EPS 1e-5f

__device__ __forceinline__ float sgn(float w) { return w >= 0.f ? 1.f : -1.f; }

__device__ __forceinline__ v16h cat8(v8h lo, v8h hi) {
    return __builtin_shufflevector(lo, hi, 0, 1, 2, 3, 4, 5, 6, 7,
                                           8, 9, 10, 11, 12, 13, 14, 15);
}

// A fragment (16x32 f16): lane needs K = ks*32 + {g*8..g*8+7} and +16.
__device__ __forceinline__ v16h load_a(const _Float16* row, int ks, int g) {
    v8h lo = *(const v8h*)(row + ks * 32 + g * 8);
    v8h hi = *(const v8h*)(row + ks * 32 + 16 + g * 8);
    return cat8(lo, hi);
}

// B fragment (32x16 f16): lane needs K = ks*32 + {g*8..g*8+7} and +16.
__device__ __forceinline__ v16h load_b(const _Float16* col, int ks, int g) {
    v8h lo = *(const v8h*)(col + ks * 32 + g * 8);
    v8h hi = *(const v8h*)(col + ks * 32 + 16 + g * 8);
    return cat8(lo, hi);
}

// ---------------------------------------------------------------------------
// Stage 1: BinarizeConv2d(1,32,(1,9),stride(1,2),pad(0,4)) + bias+BN+ReLU
//          + MaxPool(1,2).  Cin=1 -> VALU kernel. act1: [b*6+h][w32][c32] f16
// ---------------------------------------------------------------------------
__global__ void k1_conv1(const float* __restrict__ x,
                         const float* __restrict__ w1, const float* __restrict__ b1,
                         const float* __restrict__ g1, const float* __restrict__ be1,
                         const float* __restrict__ m1, const float* __restrict__ v1,
                         _Float16* __restrict__ act1) {
    int idx = blockIdx.x * blockDim.x + threadIdx.x;
    const int TOT = 4096 * 6 * 32 * 32;
    if (idx >= TOT) return;
    int c = idx & 31;
    int w = (idx >> 5) & 31;
    int rem = idx >> 10;
    int h = rem % 6;
    int b = rem / 6;

    float sw[9];
#pragma unroll
    for (int k = 0; k < 9; ++k) sw[k] = sgn(w1[c * 9 + k]);

    const float* xrow = x + (b * 6 + h) * 128;
    float bias  = b1[c];
    float scale = g1[c] * rsqrtf(v1[c] + EPS);
    float shift = be1[c] - m1[c] * scale;

    float best = 0.f;
#pragma unroll
    for (int t = 0; t < 2; ++t) {
        int wo = 2 * w + t;
        float acc = bias;
#pragma unroll
        for (int k = 0; k < 9; ++k) {
            int xw = wo * 2 + k - 4;
            if (xw >= 0 && xw < 128) acc += sw[k] * xrow[xw];
        }
        best = fmaxf(best, fmaxf(acc * scale + shift, 0.f));
    }
    act1[idx] = (_Float16)best;
}

// ---------------------------------------------------------------------------
// Stage 2: conv(32->64,(1,3),pad 1) + bias+BN+ReLU.
// Per (b,h) row: GEMM M=64 x N=32 x K=96.  4 rows per block (amortize weights).
// ---------------------------------------------------------------------------
__global__ void k2_conv2(const _Float16* __restrict__ act1,
                         const float* __restrict__ w2, const float* __restrict__ b2,
                         const float* __restrict__ g2, const float* __restrict__ be2,
                         const float* __restrict__ m2, const float* __restrict__ v2,
                         _Float16* __restrict__ act2) {
    __shared__ __align__(16) _Float16 sW[64 * 96];   // [co][K], K = ci*3+kw
    __shared__ __align__(16) _Float16 sB[32 * 96];   // [n][K] im2col tile
    __shared__ float sSc[64], sSh[64];
    int tid = threadIdx.x;

    for (int i = tid; i < 64 * 96; i += 256) {
        int co = i / 96, K = i % 96;
        int ci = K / 3, kw = K % 3;
        sW[i] = (_Float16)sgn(w2[(co * 32 + ci) * 3 + kw]);
    }
    if (tid < 64) {
        float sc = g2[tid] * rsqrtf(v2[tid] + EPS);
        sSc[tid] = sc;
        sSh[tid] = b2[tid] * sc + be2[tid] - m2[tid] * sc;  // bias folded in
    }

    int lane = tid & 31, wid = tid >> 5;
    int g = lane >> 4, ln = lane & 15;
    int wm = wid & 3;                 // co tile (4 tiles of 16)
    int wn = wid >> 2;                // w tile (2 tiles of 16)

    for (int rr = 0; rr < 4; ++rr) {
        int bh = blockIdx.x * 4 + rr;
        __syncthreads();
        for (int i = tid; i < 32 * 96; i += 256) {
            int n = i / 96, K = i % 96;
            int ci = K / 3, kw = K % 3;
            int wsrc = n + kw - 1;
            _Float16 v = (_Float16)0.f;
            if (wsrc >= 0 && wsrc < 32) v = act1[(bh * 32 + wsrc) * 32 + ci];
            sB[i] = v;
        }
        __syncthreads();

        v8f acc = {};
#pragma unroll
        for (int ks = 0; ks < 3; ++ks) {
            v16h a = load_a(&sW[(wm * 16 + ln) * 96], ks, g);
            v16h b = load_b(&sB[(wn * 16 + ln) * 96], ks, g);
            acc = __builtin_amdgcn_wmma_f32_16x16x32_f16(false, a, false, b,
                                                         (short)0, acc, false, false);
        }
        int cob = wm * 16 + g * 8;
        int w   = wn * 16 + ln;
        v8h o;
#pragma unroll
        for (int r = 0; r < 8; ++r)
            o[r] = (_Float16)fmaxf(acc[r] * sSc[cob + r] + sSh[cob + r], 0.f);
        *(v8h*)&act2[(bh * 32 + w) * 64 + cob] = o;   // 16B store, co-contiguous
    }
}

// ---------------------------------------------------------------------------
// Stage 3: conv(64->128,(1,3),pad 1) + bias+BN+ReLU + MaxPool(1,2).
// Per (b,h) row: GEMM M=128 x N=32 x K=192. 4 rows per block; pool via shfl.
// ---------------------------------------------------------------------------
__global__ void k3_conv3(const _Float16* __restrict__ act2,
                         const float* __restrict__ w3, const float* __restrict__ b3,
                         const float* __restrict__ g3, const float* __restrict__ be3,
                         const float* __restrict__ m3, const float* __restrict__ v3,
                         _Float16* __restrict__ act3) {
    __shared__ __align__(16) _Float16 sW[128 * 192]; // 48 KB, [co][K]
    __shared__ __align__(16) _Float16 sB[32 * 192];  // 12 KB, [n][K] im2col
    __shared__ float sSc[128], sSh[128];
    int tid = threadIdx.x;

    for (int i = tid; i < 128 * 192; i += 256) {
        int co = i / 192, K = i % 192;
        int ci = K / 3, kw = K % 3;
        sW[i] = (_Float16)sgn(w3[(co * 64 + ci) * 3 + kw]);
    }
    if (tid < 128) {
        float sc = g3[tid] * rsqrtf(v3[tid] + EPS);
        sSc[tid] = sc;
        sSh[tid] = b3[tid] * sc + be3[tid] - m3[tid] * sc;
    }

    int lane = tid & 31, wid = tid >> 5;   // wid = co tile (8 tiles of 16)
    int g = lane >> 4, ln = lane & 15;

    for (int rr = 0; rr < 4; ++rr) {
        int bh = blockIdx.x * 4 + rr;
        __syncthreads();
        for (int i = tid; i < 32 * 192; i += 256) {
            int n = i / 192, K = i % 192;
            int ci = K / 3, kw = K % 3;
            int wsrc = n + kw - 1;
            _Float16 v = (_Float16)0.f;
            if (wsrc >= 0 && wsrc < 32) v = act2[(bh * 32 + wsrc) * 64 + ci];
            sB[i] = v;
        }
        __syncthreads();

        for (int nt = 0; nt < 2; ++nt) {
            v8f acc = {};
#pragma unroll
            for (int ks = 0; ks < 6; ++ks) {
                v16h a = load_a(&sW[(wid * 16 + ln) * 192], ks, g);
                v16h b = load_b(&sB[(nt * 16 + ln) * 192], ks, g);
                acc = __builtin_amdgcn_wmma_f32_16x16x32_f16(false, a, false, b,
                                                             (short)0, acc, false, false);
            }
            int cob = wid * 16 + g * 8;
            v8h o;
#pragma unroll
            for (int r = 0; r < 8; ++r) {
                float val = fmaxf(acc[r] * sSc[cob + r] + sSh[cob + r], 0.f);
                o[r] = (_Float16)fmaxf(val, __shfl_xor(val, 1, 32)); // pool pair
            }
            if ((ln & 1) == 0) {
                int w2o = (nt * 16 + ln) >> 1;
                *(v8h*)&act3[(bh * 16 + w2o) * 128 + cob] = o;
            }
        }
    }
}

// ---------------------------------------------------------------------------
// Stage 4: conv(128->128,(6,1),valid) + bias+BN+ReLU.
// Per image: GEMM M=128(co) x N=16(w) x K=768(ci*6+h); weights K-chunked x128.
// The 24KB contiguous image is DMA'd into LDS by the Tensor Data Mover
// (tensor_load_to_lds, TENSORcnt), then im2col'd LDS->LDS.
// ---------------------------------------------------------------------------
__global__ void k4_conv4(const _Float16* __restrict__ act3,
                         const float* __restrict__ w4, const float* __restrict__ b4,
                         const float* __restrict__ g4, const float* __restrict__ be4,
                         const float* __restrict__ m4, const float* __restrict__ v4,
                         _Float16* __restrict__ act4) {
    __shared__ __align__(16) _Float16 sRaw[6 * 16 * 128]; // 24 KB raw [h][w][ci]
    __shared__ __align__(16) _Float16 sB[16 * 768];       // 24 KB [w][K] im2col
    __shared__ __align__(16) _Float16 sW[128 * 128];      // 32 KB, one K-chunk
    __shared__ float sSc[128], sSh[128];
    int tid = threadIdx.x;
    int b   = blockIdx.x;

    __builtin_prefetch(w4, 0, 1);                        // global_prefetch_b8

    // --- TDM: async DMA act3[b] (12288 f16, contiguous) -> sRaw -------------
    if (tid < 32) {
        unsigned lds_off =
            (unsigned)(unsigned long long)(__attribute__((address_space(3))) _Float16*)sRaw;
        unsigned long long ga =
            (unsigned long long)(const void*)(act3 + (size_t)b * 12288);
        u32x4 g0;
        g0[0] = 1u;                                       // count=1, user D#
        g0[1] = lds_off;                                  // lds_addr (bytes)
        g0[2] = (unsigned)(ga & 0xFFFFFFFFull);           // global_addr[31:0]
        g0[3] = (unsigned)((ga >> 32) & 0x1FFFFFFull)     // global_addr[56:32]
                | 0x80000000u;                            // type=2 (image)
        i32x8 g1;
        g1[0] = 0x00010000;       // workgroup_mask=0, data_size=1 (2 bytes)
        g1[1] = 0x30000000;       // tensor_dim0 = 12288 (bits 79:48 -> [31:16])
        g1[2] = 0x00010000;       // tensor_dim1 = 1 (bits 111:80 -> [31:16])
        g1[3] = 0x30000000;       // tile_dim0 = 12288 (bits 127:112)
        g1[4] = 0x00000001;       // tile_dim1 = 1
        g1[5] = 12288;            // tensor_dim0_stride[31:0]
        g1[6] = 0;
        g1[7] = 0;
        i32x4 gz  = {0, 0, 0, 0};                 // groups 2/3 unused (2-D)
        i32x8 gz8 = {0, 0, 0, 0, 0, 0, 0, 0};     // 6-arg form: extra group
        __builtin_amdgcn_tensor_load_to_lds(g0, g1, gz, gz, gz8, 0);
        __builtin_amdgcn_s_wait_tensorcnt((short)0);      // s_wait_tensorcnt 0
    }
    if (tid < 128) {
        float sc = g4[tid] * rsqrtf(v4[tid] + EPS);
        sSc[tid] = sc;
        sSh[tid] = b4[tid] * sc + be4[tid] - m4[tid] * sc;
    }
    __syncthreads();

    // im2col LDS->LDS: sB[w][ci*6+h] = sRaw[h][w][ci]
    for (int i = tid; i < 16 * 768; i += 256) {
        int n = i / 768, K = i % 768;
        int ci = K / 6, h = K % 6;
        sB[i] = sRaw[(h * 16 + n) * 128 + ci];
    }

    int lane = tid & 31, wid = tid >> 5;   // wid = co tile
    int g = lane >> 4, ln = lane & 15;
    v8f acc = {};
    for (int kc = 0; kc < 6; ++kc) {
        __syncthreads();
        for (int i = tid; i < 128 * 128; i += 256) {
            int co = i >> 7, kl = i & 127;
            int K  = kc * 128 + kl;
            int ci = K / 6, h = K % 6;
            sW[i] = (_Float16)sgn(w4[(co * 128 + ci) * 6 + h]);
        }
        __syncthreads();
#pragma unroll
        for (int ks = 0; ks < 4; ++ks) {
            v16h a = load_a(&sW[(wid * 16 + ln) * 128], ks, g);
            v16h bb = load_b(&sB[ln * 768 + kc * 128], ks, g);
            acc = __builtin_amdgcn_wmma_f32_16x16x32_f16(false, a, false, bb,
                                                         (short)0, acc, false, false);
        }
    }
#pragma unroll
    for (int r = 0; r < 8; ++r) {
        int co = wid * 16 + g * 8 + r;
        float val = fmaxf(acc[r] * sSc[co] + sSh[co], 0.f);
        act4[(size_t)b * 2048 + co * 16 + ln] = (_Float16)val;
    }
}

// ---------------------------------------------------------------------------
// Stage 5: BinarizeLinear(2048 -> 10). A = batch rows straight from global
// (K-contiguous -> global_load_b128 pairs), B = sign(wfc) staged per K-chunk.
// ---------------------------------------------------------------------------
__global__ void k5_fc(const _Float16* __restrict__ act4,
                      const float* __restrict__ wfc, const float* __restrict__ bfc,
                      float* __restrict__ out) {
    __shared__ __align__(16) _Float16 sBW[16 * 256];  // [o][k_local], 8 KB
    int tid  = threadIdx.x;
    int lane = tid & 31, wid = tid >> 5;
    int g = lane >> 4, ln = lane & 15;
    int base = (blockIdx.x * 8 + wid) * 16;

    const _Float16* arow = act4 + (size_t)(base + ln) * 2048;
    v8f acc = {};
    for (int kc = 0; kc < 8; ++kc) {
        __syncthreads();
        for (int i = tid; i < 16 * 256; i += 256) {
            int o = i >> 8, kl = i & 255;
            _Float16 v = (_Float16)0.f;
            if (o < 10) v = (_Float16)sgn(wfc[o * 2048 + kc * 256 + kl]);
            sBW[i] = v;
        }
        __syncthreads();
#pragma unroll
        for (int ks = 0; ks < 8; ++ks) {
            v8h alo = *(const v8h*)(arow + kc * 256 + ks * 32 + g * 8);
            v8h ahi = *(const v8h*)(arow + kc * 256 + ks * 32 + 16 + g * 8);
            v16h a  = cat8(alo, ahi);
            v16h bb = load_b(&sBW[ln * 256], ks, g);
            acc = __builtin_amdgcn_wmma_f32_16x16x32_f16(false, a, false, bb,
                                                         (short)0, acc, false, false);
        }
    }
#pragma unroll
    for (int r = 0; r < 8; ++r) {
        int row = base + g * 8 + r;
        if (ln < 10) out[row * 10 + ln] = acc[r] + bfc[ln];
    }
}

// ---------------------------------------------------------------------------
extern "C" void kernel_launch(void* const* d_in, const int* in_sizes, int n_in,
                              void* d_out, int out_size, void* d_ws, size_t ws_size,
                              hipStream_t stream) {
    (void)in_sizes; (void)n_in; (void)out_size; (void)ws_size;
    const float* x   = (const float*)d_in[0];
    const float* w1  = (const float*)d_in[1];
    const float* b1  = (const float*)d_in[2];
    const float* g1  = (const float*)d_in[3];
    const float* be1 = (const float*)d_in[4];
    const float* m1  = (const float*)d_in[5];
    const float* v1  = (const float*)d_in[6];
    const float* w2  = (const float*)d_in[7];
    const float* b2  = (const float*)d_in[8];
    const float* g2  = (const float*)d_in[9];
    const float* be2 = (const float*)d_in[10];
    const float* m2  = (const float*)d_in[11];
    const float* v2  = (const float*)d_in[12];
    const float* w3  = (const float*)d_in[13];
    const float* b3  = (const float*)d_in[14];
    const float* g3  = (const float*)d_in[15];
    const float* be3 = (const float*)d_in[16];
    const float* m3  = (const float*)d_in[17];
    const float* v3  = (const float*)d_in[18];
    const float* w4  = (const float*)d_in[19];
    const float* b4  = (const float*)d_in[20];
    const float* g4  = (const float*)d_in[21];
    const float* be4 = (const float*)d_in[22];
    const float* m4  = (const float*)d_in[23];
    const float* v4  = (const float*)d_in[24];
    const float* wfc = (const float*)d_in[25];
    const float* bfc = (const float*)d_in[26];
    float* out = (float*)d_out;

    _Float16* ws   = (_Float16*)d_ws;
    _Float16* act1 = ws;                     // 25,165,824 halfs (50.3 MB)
    _Float16* act2 = ws + 25165824;          // 50,331,648 halfs (100.7 MB)
    _Float16* act3 = ws + 75497472;          // 50,331,648 halfs (100.7 MB)
    _Float16* act4 = ws;                     // reuse act1 region (act1 dead)

    const int N1 = 4096 * 6 * 32 * 32;
    k1_conv1<<<(N1 + 255) / 256, 256, 0, stream>>>(x, w1, b1, g1, be1, m1, v1, act1);
    k2_conv2<<<4096 * 6 / 4, 256, 0, stream>>>(act1, w2, b2, g2, be2, m2, v2, act2);
    k3_conv3<<<4096 * 6 / 4, 256, 0, stream>>>(act2, w3, b3, g3, be3, m3, v3, act3);
    k4_conv4<<<4096, 256, 0, stream>>>(act3, w4, b4, g4, be4, m4, v4, act4);
    k5_fc<<<32, 256, 0, stream>>>(act4, wfc, bfc, out);
}
